// GliomaMRIContrastiveLoss_47373489275492
// MI455X (gfx1250) — matrix operand order repaired and checked by reading, not compiled, not run on hardware
//
#include <hip/hip_runtime.h>

// CDNA5 / gfx1250 vector types
typedef __attribute__((ext_vector_type(16))) __bf16 v16bf;
typedef __attribute__((ext_vector_type(8)))  __bf16 v8bf;
typedef __attribute__((ext_vector_type(8)))  float  v8f;
typedef __attribute__((ext_vector_type(4)))  unsigned int v4u;
typedef __attribute__((ext_vector_type(8)))  int v8i;
typedef __attribute__((ext_vector_type(4)))  int v4i;

#define NROWS 8192      // 2B
#define BHALF 4096      // B
#define DIM   256
#define INV_TEMP 2.0f   // 1 / 0.5
#define BROW  264       // LDS row stride in elements: 256 + 8 (TDM pad: 4 dwords/row)

// ---------------------------------------------------------------------------
// Kernel 1: row-normalize concat(x_i, x_j) -> bf16 zn.  One wave32 per row.
// ---------------------------------------------------------------------------
__global__ void k_normalize(const float* __restrict__ xi,
                            const float* __restrict__ xj,
                            __bf16* __restrict__ zn) {
    const int wave = (blockIdx.x * blockDim.x + threadIdx.x) >> 5;
    const int lane = threadIdx.x & 31;
    if (wave >= NROWS) return;
    const float* src = (wave < BHALF) ? (xi + (size_t)wave * DIM)
                                      : (xj + (size_t)(wave - BHALF) * DIM);
    float v[8];
    const float4 f0 = ((const float4*)src)[lane * 2 + 0];
    const float4 f1 = ((const float4*)src)[lane * 2 + 1];
    v[0]=f0.x; v[1]=f0.y; v[2]=f0.z; v[3]=f0.w;
    v[4]=f1.x; v[5]=f1.y; v[6]=f1.z; v[7]=f1.w;
    float ss = 0.f;
#pragma unroll
    for (int i = 0; i < 8; ++i) ss += v[i] * v[i];
#pragma unroll
    for (int m = 1; m < 32; m <<= 1) ss += __shfl_xor(ss, m, 32);
    const float inv = 1.0f / fmaxf(sqrtf(ss), 1e-8f);
    v8bf o;
#pragma unroll
    for (int i = 0; i < 8; ++i) o[i] = (__bf16)(v[i] * inv);
    *(v8bf*)&zn[(size_t)wave * DIM + lane * 8] = o;
}

// ---------------------------------------------------------------------------
// Kernel 2: positive-pair similarities  pos[k] = zn[k] . zn[k+B]
// ---------------------------------------------------------------------------
__global__ void k_pos(const __bf16* __restrict__ zn, float* __restrict__ pos) {
    const int wave = (blockIdx.x * blockDim.x + threadIdx.x) >> 5;
    const int lane = threadIdx.x & 31;
    if (wave >= BHALF) return;
    const v8bf a = *(const v8bf*)&zn[(size_t)wave * DIM + lane * 8];
    const v8bf b = *(const v8bf*)&zn[(size_t)(wave + BHALF) * DIM + lane * 8];
    float s = 0.f;
#pragma unroll
    for (int i = 0; i < 8; ++i) s += (float)a[i] * (float)b[i];
#pragma unroll
    for (int m = 1; m < 32; m <<= 1) s += __shfl_xor(s, m, 32);
    if (lane == 0) pos[wave] = s;
}

// ---------------------------------------------------------------------------
// Tensor Data Mover: async-load one 16-row x 256-col bf16 tile of zn into LDS.
// D# per CDNA5 ISA ch.8: 2D tensor [8192][256] of 2-byte elems, tile 256x16,
// pad 4 dwords after every 128 dwords (one row) -> LDS row stride 528 B.
// ---------------------------------------------------------------------------
__device__ __forceinline__ void tdm_load_tile(const __bf16* gsrc, unsigned lds_off) {
    const unsigned long long ga = (unsigned long long)(size_t)gsrc;
    v4u g0;
    g0[0] = 1u;                                            // count=1, user desc
    g0[1] = lds_off;                                       // lds_addr (bytes)
    g0[2] = (unsigned)(ga & 0xFFFFFFFFu);                  // global_addr[31:0]
    g0[3] = (unsigned)((ga >> 32) & 0x01FFFFFFu)           // global_addr[56:32]
          | 0x80000000u;                                   // type=2 ("image")
    v8i g1;
    g1[0] = (int)((1u << 16)    // data_size = 1 -> 2 bytes
                | (1u << 20)    // pad_enable
                | (6u << 22)    // pad_interval code 6 -> every 128 dwords
                | (3u << 25));  // pad_amount  code 3 -> 4 dwords
    g1[1] = (int)(256u << 16);  // tensor_dim0 = 256 (low 16 bits)
    g1[2] = (int)(8192u << 16); // tensor_dim1 = 8192 (low 16 bits)
    g1[3] = (int)(256u << 16);  // tile_dim0 = 256
    g1[4] = 16;                 // tile_dim1 = 16, tile_dim2 = 0
    g1[5] = 256;                // tensor_dim0_stride[31:0] = 256 elements
    g1[6] = 0;                  // dim0_stride[47:32]=0, dim1_stride[15:0]=0
    g1[7] = 0;
    const v4i z4 = {0, 0, 0, 0};
#if __has_include(<hip/amd_detail/amd_gfx1250_TDM.h>)
    const v8i z8 = {0, 0, 0, 0, 0, 0, 0, 0};
    __builtin_amdgcn_tensor_load_to_lds(g0, g1, z4, z4, z8, 0);
#else
    __builtin_amdgcn_tensor_load_to_lds(g0, g1, z4, z4, 0);
#endif
}

// ---------------------------------------------------------------------------
// Kernel 3: WMMA Gram tiles fused with masked exp-row-sum.
// Block = 128 threads (4 waves), workgroup owns 64 rows, wave owns 16 rows.
// A fragments hoisted to VGPRs (loop-invariant); B tiles double-buffered in
// LDS via the Tensor Data Mover (TENSORcnt), one barrier per column tile.
// ---------------------------------------------------------------------------
__global__ void __launch_bounds__(128)
k_sim_expsum(const __bf16* __restrict__ zn, float* __restrict__ denom) {
    const int tid  = threadIdx.x;      // 0..127
    const int wave = tid >> 5;         // 0..3
    const int lane = tid & 31;
    const int hi   = lane >> 4;        // half-wave select
    const int rowBase = blockIdx.x * 64;

    __shared__ __bf16 Bbuf[2][16 * BROW];   // 2 x 8448 B, TDM-padded rows

    // --- Hoist A fragments: 16 rows x 256 K per wave, straight from global
    // (zn is L2-resident).  ISA 16-bit A 16x32 layout: lo lanes K{0-7,16-23},
    // hi lanes K{8-15,24-31} for each 32-wide K step.
    const int aRow = rowBase + wave * 16 + (lane & 15);
    v16bf afrag[8];
#pragma unroll
    for (int kk = 0; kk < 8; ++kk) {
        const __bf16* ap = &zn[(size_t)aRow * DIM + kk * 32 + hi * 8];
        const v8bf a0 = *(const v8bf*)(ap + 0);
        const v8bf a1 = *(const v8bf*)(ap + 16);
#pragma unroll
        for (int i = 0; i < 8; ++i) { afrag[kk][i] = a0[i]; afrag[kk][i + 8] = a1[i]; }
    }

    float den[8];
#pragma unroll
    for (int r = 0; r < 8; ++r) den[r] = 0.f;

    const unsigned lds0 = (unsigned)(size_t)(const void*)&Bbuf[0][0];
    const unsigned lds1 = (unsigned)(size_t)(const void*)&Bbuf[1][0];

    // Prologue: wave 0 kicks off the DMA for tile 0.
    if (wave == 0) tdm_load_tile(zn, lds0);

    for (int ct = 0; ct < NROWS / 16; ++ct) {
        if (wave == 0) __builtin_amdgcn_s_wait_tensorcnt(0); // tile ct landed
        __syncthreads();   // tile ct visible; buf[(ct+1)&1] no longer read
        if (wave == 0 && ct + 1 < NROWS / 16)
            tdm_load_tile(zn + (size_t)(ct + 1) * 16 * DIM,
                          ((ct + 1) & 1) ? lds1 : lds0);

        // B fragment base: lane's sim-column = zn row (colBase + lane%16),
        // padded LDS row stride BROW keeps half-wave b128 reads conflict-free.
        const __bf16* bbase = &Bbuf[ct & 1][(lane & 15) * BROW + hi * 8];

        v8f c = {0.f, 0.f, 0.f, 0.f, 0.f, 0.f, 0.f, 0.f};
#pragma unroll
        for (int kk = 0; kk < 8; ++kk) {
            const __bf16* bp = bbase + kk * 32;
            const v8bf b0 = *(const v8bf*)(bp + 0);
            const v8bf b1 = *(const v8bf*)(bp + 16);
            v16bf b;
#pragma unroll
            for (int i = 0; i < 8; ++i) { b[i] = b0[i]; b[i + 8] = b1[i]; }
            c = __builtin_amdgcn_wmma_f32_16x16x32_bf16(
                    false, afrag[kk], false, b, (short)0, c, false, false);
        }

        // Fused epilogue: exp(sim/T), diagonal masked, accumulate in VGPRs.
        const int gCol = ct * 16 + (lane & 15);
#pragma unroll
        for (int r = 0; r < 8; ++r) {
            const int gRow = rowBase + wave * 16 + r + 8 * hi;
            const float e = __expf(c[r] * INV_TEMP);
            den[r] += (gRow != gCol) ? e : 0.0f;
        }
    }

    // Reduce each row's partials across its 16 lanes (xor<16 stays in half).
#pragma unroll
    for (int r = 0; r < 8; ++r) {
        float v = den[r];
#pragma unroll
        for (int m = 1; m < 16; m <<= 1) v += __shfl_xor(v, m, 32);
        if ((lane & 15) == 0)
            denom[rowBase + wave * 16 + r + 8 * hi] = v;
    }
}

// ---------------------------------------------------------------------------
// Kernel 4: loss = mean over 2B rows of (log(denom) - pos/T)
// ---------------------------------------------------------------------------
__global__ void k_finalize(const float* __restrict__ denom,
                           const float* __restrict__ pos,
                           float* __restrict__ out) {
    __shared__ float red[1024];
    float s = 0.f;
    for (int r = threadIdx.x; r < NROWS; r += 1024)
        s += logf(denom[r]) - pos[r & (BHALF - 1)] * INV_TEMP;
    red[threadIdx.x] = s;
    __syncthreads();
    for (int off = 512; off > 0; off >>= 1) {
        if (threadIdx.x < off) red[threadIdx.x] += red[threadIdx.x + off];
        __syncthreads();
    }
    if (threadIdx.x == 0) out[0] = red[0] / (float)NROWS;
}

// ---------------------------------------------------------------------------
extern "C" void kernel_launch(void* const* d_in, const int* in_sizes, int n_in,
                              void* d_out, int out_size, void* d_ws, size_t ws_size,
                              hipStream_t stream) {
    const float* xi = (const float*)d_in[0];
    const float* xj = (const float*)d_in[1];
    float* out = (float*)d_out;

    char* ws = (char*)d_ws;
    __bf16* zn  = (__bf16*)ws;                                   // 4 MiB
    float*  den = (float*)(ws + (size_t)NROWS * DIM * 2);        // 32 KiB
    float*  pos = (float*)(ws + (size_t)NROWS * DIM * 2 + NROWS * 4);

    k_normalize<<<NROWS / 8, 256, 0, stream>>>(xi, xj, zn);
    k_pos<<<BHALF / 8, 256, 0, stream>>>(zn, pos);
    k_sim_expsum<<<NROWS / 64, 128, 0, stream>>>(zn, den);
    k_finalize<<<1, 1024, 0, stream>>>(den, pos, out);
}
